// mixerLayer_70935679861135
// MI455X (gfx1250) — matrix operand order repaired and checked by reading, not compiled, not run on hardware
//
#include <hip/hip_runtime.h>

// Problem constants (B=4, M=8, N=128, D=32, H=64)
#define NSETS 32
#define NN    128
#define DD    32
#define HH    64
#define TWOD  64

typedef __bf16 bf16;
typedef __attribute__((ext_vector_type(16))) __bf16 v16bf;
typedef __attribute__((ext_vector_type(8)))  float  v8f;

__device__ __forceinline__ bf16 f2bf(float f) {
  union { float f; unsigned u; } c; c.f = f;
  unsigned r = c.u + 0x7FFFu + ((c.u >> 16) & 1u);   // round-to-nearest-even
  unsigned short h = (unsigned short)(r >> 16);
  return __builtin_bit_cast(bf16, h);
}
__device__ __forceinline__ float bf2f(bf16 h) {
  unsigned short u = __builtin_bit_cast(unsigned short, h);
  unsigned x = ((unsigned)u) << 16;
  return __builtin_bit_cast(float, x);
}

// gelu (tanh variant) rewritten as x * sigmoid(2u); uses fast v_exp_f32 path.
// 2u = 2*sqrt(2/pi)*(x + 0.044715 x^3)
__device__ __forceinline__ float gelu_f(float x) {
  float t = x * (-1.5957691216057308f) - (x * x * x) * 0.07135481627362202f;
  return x / (1.0f + __expf(t));
}

// --- WMMA fragment loaders (wave32, 16x16x32 bf16, contiguous k = t + 16*g per lane half) ---

// A fragment: 16(M) x 32(K) tile from row-major bf16 LDS, row stride ld.
// lane: M = lane%16, group g = lane/16; half t -> K = t + 16*g
__device__ __forceinline__ v16bf load_a(const bf16* base, int ld, int row0, int col0, int lane) {
  int g = lane >> 4, m = lane & 15;
  const bf16* p = base + (row0 + m) * ld + col0;
  v16bf a;
#pragma unroll
  for (int t = 0; t < 16; ++t) {
    int k = t + g * 16;
    a[t] = p[k];
  }
  return a;
}

// B fragment: 32(K) x 16(N) tile from row-major bf16 LDS.
// lane: N = lane%16, g = lane/16; half t -> K = t + 16*g
__device__ __forceinline__ v16bf load_b(const bf16* base, int ld, int row0, int col0, int lane) {
  int g = lane >> 4, n = lane & 15;
  const bf16* p = base + (row0 + g * 16) * ld + col0 + n;
  v16bf b;
#pragma unroll
  for (int t = 0; t < 16; ++t) b[t] = p[t * ld];
  return b;
}

// B fragment streamed from global f32 (weights; L2-resident) with bf16 convert.
__device__ __forceinline__ v16bf load_b_gf(const float* base, int ld, int row0, int col0, int lane) {
  int g = lane >> 4, n = lane & 15;
  const float* p = base + (row0 + g * 16) * ld + col0 + n;
  v16bf b;
#pragma unroll
  for (int t = 0; t < 16; ++t) b[t] = f2bf(p[t * ld]);
  return b;
}

__device__ __forceinline__ v8f wmma_bf16(v16bf a, v16bf b, v8f c) {
  return __builtin_amdgcn_wmma_f32_16x16x32_bf16(false, a, false, b, (short)0, c, false, false);
}

__global__ __launch_bounds__(256)
void set_mixer_kernel(const float* __restrict__ x,   const float* __restrict__ xsz,
                      const float* __restrict__ W1a, const float* __restrict__ b1a,
                      const float* __restrict__ W1b, const float* __restrict__ b1b,
                      const float* __restrict__ W2a, const float* __restrict__ b2a,
                      const float* __restrict__ W2b, const float* __restrict__ b2b,
                      const float* __restrict__ w3,  const float* __restrict__ b3,
                      float* __restrict__ out)
{
  // ~57 KB static LDS
  __shared__ union {
    bf16 hb[NN][TWOD];       // MLP1 hidden (stage B only)
    bf16 PQ[2][NN][TWOD];    // P = x1@W2a + b2a, Q = xn@W2a (stages C/D)
  } big;                                  // 32 KB
  __shared__ bf16 xnb[NN][DD];            // 8 KB
  __shared__ bf16 x1b[NN][DD];            // 8 KB
  __shared__ bf16 STt[NN][32];            // 8 KB  S^T K-tile: STt[j][i_local]
  __shared__ float vvec[HH];
  __shared__ float mask_s[NN];
  __shared__ float sred[NN];
  __shared__ float stats[2];
  __shared__ float cval[1];

  const int tid  = threadIdx.x;
  const int lane = tid & 31;
  const int wave = tid >> 5;
  const int sb   = blockIdx.x;            // set index in [0, B*M)
  const int bidx = sb >> 3;               // batch index (M = 8)
  const float* xg = x + (size_t)sb * (NN * DD);

  // --- precompute v = W2b @ w3 (collapses MLP2's second GEMM) and c = b2b.w3
  if (tid < HH) {
    float acc = 0.f;
#pragma unroll 4
    for (int h = 0; h < HH; ++h) acc += W2b[tid * HH + h] * w3[h];
    vvec[tid] = acc;
  } else if (tid == HH) {
    float acc = 0.f;
    for (int h = 0; h < HH; ++h) acc += b2b[h] * w3[h];
    cval[0] = acc;
  }

  // --- stage A: masked set-norm over (N*D), denom = x_size*D ---
  float row[DD];
  if (tid < NN) {
    bool any = false;
    float rowsum = 0.f;
#pragma unroll
    for (int d = 0; d < DD; ++d) {
      float v = xg[tid * DD + d];
      row[d] = v; rowsum += v;
      any = any || (v != 0.f);
    }
    mask_s[tid] = any ? 1.f : 0.f;
    sred[tid] = rowsum;
  }
  __syncthreads();
  const float denom = xsz[bidx] * (float)DD;
  if (tid == 0) {
    float s = 0.f;
    for (int k = 0; k < NN; ++k) s += sred[k];
    stats[0] = s / denom;
  }
  __syncthreads();
  const float mean = stats[0];
  if (tid < NN) {
    float vs = 0.f;
#pragma unroll
    for (int d = 0; d < DD; ++d) { float df = row[d] - mean; vs += df * df; }
    sred[tid] = vs * mask_s[tid];
  }
  __syncthreads();
  if (tid == 0) {
    float s = 0.f;
    for (int k = 0; k < NN; ++k) s += sred[k];
    stats[1] = sqrtf(s / denom);
  }
  __syncthreads();
  if (tid < NN) {
    float inv = mask_s[tid] / (stats[1] + 1e-8f);
#pragma unroll
    for (int d = 0; d < DD; ++d) xnb[tid][d] = f2bf((row[d] - mean) * inv);
  }
  __syncthreads();

  // --- stage B1: h = gelu(xn @ W1a + b1a)  [128x64], 32 tiles, 4/wave ---
  {
#pragma unroll
    for (int q = 0; q < 4; ++q) {
      int tt = wave * 4 + q;
      int mi = tt >> 2, nj = tt & 3;
      v16bf a  = load_a(&xnb[0][0], DD, mi * 16, 0, lane);
      v16bf bb = load_b_gf(W1a, TWOD, 0, nj * 16, lane);
      v8f acc = {};
      acc = wmma_bf16(a, bb, acc);
      int n = nj * 16 + (lane & 15);
      int g = lane >> 4;
      float bias = b1a[n];
#pragma unroll
      for (int r = 0; r < 8; ++r) {
        int m = mi * 16 + r + g * 8;
        big.hb[m][n] = f2bf(gelu_f(acc[r] + bias));
      }
    }
  }
  __syncthreads();

  // --- stage B2: x1 = (h @ W1b + b1b) * mask  [128x32], 16 tiles, 2/wave ---
  {
#pragma unroll
    for (int q = 0; q < 2; ++q) {
      int tt = wave * 2 + q;
      int mi = tt >> 1, nj = tt & 1;
      v8f acc = {};
#pragma unroll
      for (int k0 = 0; k0 < TWOD; k0 += 32) {
        v16bf a  = load_a(&big.hb[0][0], TWOD, mi * 16, k0, lane);
        v16bf bb = load_b_gf(W1b, DD, k0, nj * 16, lane);
        acc = wmma_bf16(a, bb, acc);
      }
      int n = nj * 16 + (lane & 15);
      int g = lane >> 4;
      float bias = b1b[n];
#pragma unroll
      for (int r = 0; r < 8; ++r) {
        int m = mi * 16 + r + g * 8;
        x1b[m][n] = f2bf((acc[r] + bias) * mask_s[m]);
      }
    }
  }
  __syncthreads();   // hb dead; big region reused for P/Q below

  // --- stage C: P = x1@W2a + b2a, Q = xn@W2a  [128x64] each, 8 tiles/wave ---
  {
#pragma unroll
    for (int q = 0; q < 8; ++q) {
      int t = wave * 8 + q;          // 0..63; first 32 -> P, last 32 -> Q
      int isQ = t >> 5;
      int u = t & 31;
      int mi = u >> 2, nj = u & 3;
      const bf16* Asrc = isQ ? &xnb[0][0] : &x1b[0][0];
      v16bf a  = load_a(Asrc, DD, mi * 16, 0, lane);
      v16bf bb = load_b_gf(W2a, TWOD, 0, nj * 16, lane);
      v8f acc = {};
      acc = wmma_bf16(a, bb, acc);
      int n = nj * 16 + (lane & 15);
      int g = lane >> 4;
      float bias = isQ ? 0.f : b2a[n];
#pragma unroll
      for (int r = 0; r < 8; ++r) {
        int m = mi * 16 + r + g * 8;
        big.PQ[isQ][m][n] = f2bf(acc[r] + bias);
      }
    }
  }
  __syncthreads();

  // --- stage D: fused S-tile compute + out = S^T @ x1 accumulation ---
  // s[i,j] = sum_k gelu(P[i,k]-Q[j,k]) * v[k] + c ; out[j,d] = sum_i s[i,j]*x1[i,d]
  const int j0  = (tid >> 3) * 4;   // 4x4 patch per thread: 32 j-patches x 8 i-patches
  const int il0 = (tid & 7) * 4;
  const float cc = cval[0];
  const int jb = wave * 16;         // each wave owns j-strip [jb, jb+16), both d-tiles
  v8f oacc0 = {}, oacc1 = {};

  for (int it2 = 0; it2 < 4; ++it2) {   // K-tiles of 32 over i
    const int ibase = it2 * 32;
    float acc[4][4];
#pragma unroll
    for (int jj = 0; jj < 4; ++jj)
#pragma unroll
      for (int ii = 0; ii < 4; ++ii) acc[jj][ii] = cc;

    for (int k = 0; k < TWOD; ++k) {
      float vk = vvec[k];
      float pv[4], qv[4];
#pragma unroll
      for (int ii = 0; ii < 4; ++ii) pv[ii] = bf2f(big.PQ[0][ibase + il0 + ii][k]);
#pragma unroll
      for (int jj = 0; jj < 4; ++jj) qv[jj] = bf2f(big.PQ[1][j0 + jj][k]);
#pragma unroll
      for (int jj = 0; jj < 4; ++jj)
#pragma unroll
        for (int ii = 0; ii < 4; ++ii)
          acc[jj][ii] += gelu_f(pv[ii] - qv[jj]) * vk;
    }
#pragma unroll
    for (int jj = 0; jj < 4; ++jj)
#pragma unroll
      for (int ii = 0; ii < 4; ++ii)
        STt[j0 + jj][il0 + ii] = f2bf(acc[jj][ii]);
    __syncthreads();

    // accumulate out tiles: A = S^T strip [16j x 32i], B = x1 [32i x 16d]
    {
      v16bf a0  = load_a(&STt[0][0], 32, jb, 0, lane);
      v16bf bb0 = load_b(&x1b[0][0], DD, ibase, 0, lane);
      v16bf bb1 = load_b(&x1b[0][0], DD, ibase, 16, lane);
      oacc0 = wmma_bf16(a0, bb0, oacc0);
      oacc1 = wmma_bf16(a0, bb1, oacc1);
    }
    __syncthreads();
  }

  // --- epilogue: out = (S^T@x1 + b3 + x_orig) * mask ---
  {
    const float b3v = b3[0];
    int n = lane & 15, g = lane >> 4;
    float* og = out + (size_t)sb * (NN * DD);
#pragma unroll
    for (int r = 0; r < 8; ++r) {
      int j = jb + r + g * 8;
      float mk = mask_s[j];
      og[j * DD + n]      = (oacc0[r] + b3v + xg[j * DD + n])      * mk;
      og[j * DD + 16 + n] = (oacc1[r] + b3v + xg[j * DD + 16 + n]) * mk;
    }
  }
}

extern "C" void kernel_launch(void* const* d_in, const int* in_sizes, int n_in,
                              void* d_out, int out_size, void* d_ws, size_t ws_size,
                              hipStream_t stream) {
  (void)in_sizes; (void)n_in; (void)d_ws; (void)ws_size; (void)out_size;
  const float* x   = (const float*)d_in[0];
  const float* xsz = (const float*)d_in[1];
  const float* W1a = (const float*)d_in[2];
  const float* b1a = (const float*)d_in[3];
  const float* W1b = (const float*)d_in[4];
  const float* b1b = (const float*)d_in[5];
  const float* W2a = (const float*)d_in[6];
  const float* b2a = (const float*)d_in[7];
  const float* W2b = (const float*)d_in[8];
  const float* b2b = (const float*)d_in[9];
  const float* w3  = (const float*)d_in[10];
  const float* b3  = (const float*)d_in[11];
  float* outp = (float*)d_out;

  set_mixer_kernel<<<dim3(NSETS), dim3(256), 0, stream>>>(
      x, xsz, W1a, b1a, W1b, b1b, W2a, b2a, W2b, b2b, w3, b3, outp);
}
